// AttnDecoderGRU1L_SMAXOP_12369505813152
// MI455X (gfx1250) — matrix operand order repaired and checked by reading, not compiled, not run on hardware
//
#include <hip/hip_runtime.h>
#include <hip/hip_bf16.h>
#include <math.h>

// ---------------------------------------------------------------------------
// AttnDecoder GRU single step, MI455X (gfx1250, wave32).
//   Dominant GEMM (E[4096,2048] @ attn2_w^T[2048,2048]) -> v_wmma_f32_16x16x32_bf16
//   with a fused tanh/dot epilogue producing gamma[4096] directly (no 32MB uah).
//   BK=64: 4 WMMAs per wave per barrier pair; global_prefetch for next K tiles.
//   All batch-1 GEMVs are bandwidth-bound -> wave-per-row float4 streaming.
// ---------------------------------------------------------------------------

typedef __attribute__((ext_vector_type(16))) __bf16       v16bf;
typedef __attribute__((ext_vector_type(8)))  float        f32x8;
typedef __attribute__((ext_vector_type(4)))  unsigned int u32x4;

#define H1_  512
#define H_   2048
#define OUT_ 1024
#define T_   4096

#define BM 64
#define BN 64
#define BK 64

__device__ __forceinline__ unsigned short f2bf(float f) {
    unsigned int u = __float_as_uint(f);
    u += 0x7FFFu + ((u >> 16) & 1u);          // round-to-nearest-even
    return (unsigned short)(u >> 16);
}

// ---------------------------------------------------------------------------
// Fused: gamma[t] += sum_n tanh(way[n] + (E @ W2^T)[t,n]) * a3[n]
// Block = 256 threads = 8 waves, block tile 64x64, K-step 64 (2x bf16 WMMA K=32).
// ---------------------------------------------------------------------------
__global__ __launch_bounds__(256) void uah_gamma_wmma_kernel(
    const float* __restrict__ E,    // [T_, H_] row-major
    const float* __restrict__ W2,   // [H_, H_] row-major (attn2_w)
    const float* __restrict__ way,  // [H_]
    const float* __restrict__ a3,   // [H_]   (attn3_w)
    float* __restrict__ gamma)      // [T_]   pre-zeroed
{
    __shared__ unsigned short As[BM][BK];   // bf16 bits: E tile, As[m][k]
    __shared__ unsigned short Bs[BN][BK];   // bf16 bits: Bs[n][k] = W2[n0+n][k0+k]

    const int tid  = threadIdx.x;
    const int wave = tid >> 5;
    const int lane = tid & 31;
    const int m0   = blockIdx.x * BM;       // t block
    const int n0   = blockIdx.y * BN;       // h block

    // wave -> two adjacent 16x16 subtiles (same row band)
    const int sm  = wave >> 1;              // 0..3
    const int sn0 = (wave & 1) * 2;         // 0 or 2

    f32x8 acc0 = {};
    f32x8 acc1 = {};

    // cooperative tile loader: 256 threads x 4 float4 per tile per K-step
    const int lrow = tid >> 4;              // 0..15
    const int lcol = (tid & 15) * 4;        // 0,4,...,60

    for (int k0 = 0; k0 < H_; k0 += BK) {
#pragma unroll
        for (int p = 0; p < 4; ++p) {
            const int r = lrow + p * 16;
            const float* pa = E  + (size_t)(m0 + r) * H_ + k0 + lcol;
            const float* pb = W2 + (size_t)(n0 + r) * H_ + k0 + lcol;
            const float4 va = *(const float4*)pa;
            As[r][lcol + 0] = f2bf(va.x);
            As[r][lcol + 1] = f2bf(va.y);
            As[r][lcol + 2] = f2bf(va.z);
            As[r][lcol + 3] = f2bf(va.w);
            const float4 vb = *(const float4*)pb;
            Bs[r][lcol + 0] = f2bf(vb.x);
            Bs[r][lcol + 1] = f2bf(vb.y);
            Bs[r][lcol + 2] = f2bf(vb.z);
            Bs[r][lcol + 3] = f2bf(vb.w);
            // pull next K-step's tiles toward the WGP (global_prefetch_b8)
            if (k0 + BK < H_) {
                __builtin_prefetch(pa + BK, 0, 1);
                __builtin_prefetch(pb + BK, 0, 1);
            }
        }
        __syncthreads();

        const int arow  = sm * 16 + (lane & 15);
        const int khalf = (lane >> 4) & 1;
        const int bnrow0 = (sn0 + 0) * 16 + (lane & 15);
        const int bnrow1 = (sn0 + 1) * 16 + (lane & 15);
        const int klo   = (lane >> 4) * 16;

#pragma unroll
        for (int kk = 0; kk < BK; kk += 32) {
            // A fragment (16-bit A 16x32: lanes 0-15 K0..15, lanes 16-31 K16..31)
            union { v16bf v; u32x4 q[2]; } ua;
            ua.q[0] = *(const u32x4*)&As[arow][kk + khalf * 16];
            ua.q[1] = *(const u32x4*)&As[arow][kk + khalf * 16 + 8];
            // B fragments (32x16 B: lanes 0-15 => N=lane,K0..15; lanes 16-31 => N=lane-16,K16..31)
            union { v16bf v; u32x4 q[2]; } ub0, ub1;
            const u32x4* p0 = (const u32x4*)&Bs[bnrow0][kk + klo];
            ub0.q[0] = p0[0]; ub0.q[1] = p0[1];
            const u32x4* p1 = (const u32x4*)&Bs[bnrow1][kk + klo];
            ub1.q[0] = p1[0]; ub1.q[1] = p1[1];

            acc0 = __builtin_amdgcn_wmma_f32_16x16x32_bf16(
                false, ua.v, false, ub0.v, (short)0, acc0, false, false);
            acc1 = __builtin_amdgcn_wmma_f32_16x16x32_bf16(
                false, ua.v, false, ub1.v, (short)0, acc1, false, false);
        }
        __syncthreads();
    }

    // Epilogue: C layout lane<16 => N=lane,M=r ; lane>=16 => N=lane-16,M=r+8.
    // gamma[t] += sum over this tile's 16 columns of tanh(way+uah)*a3.
    const int half = lane >> 4;
    const int nl   = lane & 15;
#pragma unroll
    for (int which = 0; which < 2; ++which) {
        const f32x8 acc = which ? acc1 : acc0;
        const int   n   = n0 + (sn0 + which) * 16 + nl;
        const float wv  = way[n];
        const float av  = a3[n];
#pragma unroll
        for (int r = 0; r < 8; ++r) {
            float c = tanhf(wv + acc[r]) * av;
            c += __shfl_xor(c, 1, 32);      // xor<=8 stays inside each 16-lane half
            c += __shfl_xor(c, 2, 32);
            c += __shfl_xor(c, 4, 32);
            c += __shfl_xor(c, 8, 32);
            if (nl == 0)
                atomicAdd(&gamma[m0 + sm * 16 + r + 8 * half], c);
        }
    }
}

// ---------------------------------------------------------------------------
// Wave-per-row GEMV: y = act(W @ x + b), W row-major [rows,k], k % 4 == 0.
// ---------------------------------------------------------------------------
__global__ __launch_bounds__(256) void gemv_kernel(
    const float* __restrict__ W, const float* __restrict__ x,
    const float* __restrict__ b, float* __restrict__ y,
    int rows, int k, int relu)
{
    const int wv   = blockIdx.x * (blockDim.x >> 5) + (threadIdx.x >> 5);
    const int lane = threadIdx.x & 31;
    if (wv >= rows) return;
    const float4* w4 = (const float4*)(W + (size_t)wv * k);
    const float4* x4 = (const float4*)x;
    const int k4 = k >> 2;
    float acc = 0.f;
    for (int i = lane; i < k4; i += 32) {
        const float4 a = w4[i];
        const float4 c = x4[i];
        acc += a.x * c.x + a.y * c.y + a.z * c.z + a.w * c.w;
    }
    acc += __shfl_xor(acc, 16, 32);
    acc += __shfl_xor(acc, 8, 32);
    acc += __shfl_xor(acc, 4, 32);
    acc += __shfl_xor(acc, 2, 32);
    acc += __shfl_xor(acc, 1, 32);
    if (lane == 0) {
        if (b) acc += b[wv];
        if (relu) acc = fmaxf(acc, 0.f);
        y[wv] = acc;
    }
}

// out[h] += sum_{t in chunk} w[t] * E[t,h]   (coalesced over h; atomic partials)
__global__ __launch_bounds__(256) void attn_apply_kernel(
    const float* __restrict__ w, const float* __restrict__ E,
    float* __restrict__ out)
{
    const int h  = blockIdx.x * 256 + threadIdx.x;
    const int t0 = blockIdx.y * 256;
    float acc = 0.f;
    for (int t = t0; t < t0 + 256; ++t)
        acc += w[t] * E[(size_t)t * H_ + h];
    atomicAdd(&out[h], acc);
}

// single-block softmax, n <= 4096, 1024 threads
__global__ __launch_bounds__(1024) void softmax_kernel(
    const float* __restrict__ x, float* __restrict__ y, int n)
{
    __shared__ float red[33];
    const int tid  = threadIdx.x;
    const int wid  = tid >> 5;
    const int lane = tid & 31;
    const int nw   = blockDim.x >> 5;

    float m = -3.402823e38f;
    for (int i = tid; i < n; i += blockDim.x) m = fmaxf(m, x[i]);
    for (int o = 16; o >= 1; o >>= 1) m = fmaxf(m, __shfl_xor(m, o, 32));
    if (lane == 0) red[wid] = m;
    __syncthreads();
    if (wid == 0) {
        float v = (lane < nw) ? red[lane] : -3.402823e38f;
        for (int o = 16; o >= 1; o >>= 1) v = fmaxf(v, __shfl_xor(v, o, 32));
        if (lane == 0) red[32] = v;
    }
    __syncthreads();
    m = red[32];
    __syncthreads();

    float s = 0.f;
    for (int i = tid; i < n; i += blockDim.x) {
        const float e = expf(x[i] - m);
        y[i] = e;
        s += e;
    }
    for (int o = 16; o >= 1; o >>= 1) s += __shfl_xor(s, o, 32);
    if (lane == 0) red[wid] = s;
    __syncthreads();
    if (wid == 0) {
        float v = (lane < nw) ? red[lane] : 0.f;
        for (int o = 16; o >= 1; o >>= 1) v += __shfl_xor(v, o, 32);
        if (lane == 0) red[32] = v;
    }
    __syncthreads();
    const float inv = 1.f / red[32];
    for (int i = tid; i < n; i += blockDim.x) y[i] *= inv;
}

__global__ __launch_bounds__(256) void concat_kernel(
    const float* __restrict__ a, const float* __restrict__ b,
    float* __restrict__ x, int n)
{
    const int i = blockIdx.x * 256 + threadIdx.x;
    if (i < n)            x[i] = a[i];
    else if (i < 2 * n)   x[i] = b[i - n];
}

// PyTorch gate order r,z,n
__global__ __launch_bounds__(256) void gru_combine_kernel(
    const float* __restrict__ gi, const float* __restrict__ gh,
    const float* __restrict__ h0, float* __restrict__ h_new)
{
    const int h = blockIdx.x * 256 + threadIdx.x;
    if (h >= H_) return;
    const float r = 1.f / (1.f + expf(-(gi[h]          + gh[h])));
    const float z = 1.f / (1.f + expf(-(gi[H_ + h]     + gh[H_ + h])));
    const float n = tanhf(gi[2 * H_ + h] + r * gh[2 * H_ + h]);
    h_new[h] = (1.f - z) * n + z * h0[h];
}

// ---------------------------------------------------------------------------
extern "C" void kernel_launch(void* const* d_in, const int* in_sizes, int n_in,
                              void* d_out, int out_size, void* d_ws, size_t ws_size,
                              hipStream_t stream)
{
    (void)in_sizes; (void)n_in; (void)out_size; (void)ws_size;

    const float* input     = (const float*)d_in[0];
    const float* hidden    = (const float*)d_in[1];   // [1,1,H] -> flat [H]
    const float* E         = (const float*)d_in[2];   // [T,H]
    const float* prenet_w  = (const float*)d_in[3];
    const float* prenet_b  = (const float*)d_in[4];
    const float* prenet2_w = (const float*)d_in[5];
    const float* prenet2_b = (const float*)d_in[6];
    const float* attn1_w   = (const float*)d_in[7];
    const float* attn2_w   = (const float*)d_in[8];
    const float* attn3_w   = (const float*)d_in[9];
    const float* gru_w_ih  = (const float*)d_in[10];
    const float* gru_w_hh  = (const float*)d_in[11];
    const float* gru_b_ih  = (const float*)d_in[12];
    const float* gru_b_hh  = (const float*)d_in[13];
    const float* lin_w     = (const float*)d_in[14];
    const float* lin_b     = (const float*)d_in[15];

    float* out = (float*)d_out;
    float* o_softmax = out;              // [OUT_]
    float* o_hnew    = out + OUT_;       // [H_]
    float* o_attnw   = out + OUT_ + H_;  // [T_]

    float* ws    = (float*)d_ws;
    float* em6   = ws;            // 2048
    float* h1    = ws + 2048;     // 512
    float* way   = ws + 2560;     // 2048
    float* gamma = ws + 4608;     // 4096
    float* aapp  = ws + 8704;     // 2048
    float* xcat  = ws + 10752;    // 4096
    float* gi    = ws + 14848;    // 6144
    float* gh    = ws + 20992;    // 6144
    float* logit = ws + 27136;    // 1024

    hipMemsetAsync(gamma, 0, T_ * sizeof(float), stream);
    hipMemsetAsync(aapp,  0, H_ * sizeof(float), stream);

    // prenet: h1 = relu(prenet_w @ input + b) ; em6 = relu(prenet2_w @ h1 + b)
    gemv_kernel<<<H1_ / 8, 256, 0, stream>>>(prenet_w,  input, prenet_b,  h1,  H1_, OUT_, 1);
    gemv_kernel<<<H_  / 8, 256, 0, stream>>>(prenet2_w, h1,    prenet2_b, em6, H_,  H1_,  1);

    // way = attn1_w @ em6
    gemv_kernel<<<H_ / 8, 256, 0, stream>>>(attn1_w, em6, nullptr, way, H_, H_, 0);

    // fused uah GEMM (WMMA bf16) + tanh + dot(a3) -> gamma
    uah_gamma_wmma_kernel<<<dim3(T_ / BM, H_ / BN), 256, 0, stream>>>(
        E, attn2_w, way, attn3_w, gamma);

    // attn_weights = softmax(gamma)  (final output region, also consumed below)
    softmax_kernel<<<1, 1024, 0, stream>>>(gamma, o_attnw, T_);

    // attn_applied = attn_weights @ E
    attn_apply_kernel<<<dim3(H_ / 256, T_ / 256), 256, 0, stream>>>(o_attnw, E, aapp);

    // x = [em6, attn_applied]
    concat_kernel<<<(2 * H_) / 256, 256, 0, stream>>>(em6, aapp, xcat, H_);

    // GRU gates (bandwidth-dominant GEMVs: 151 MB of weights)
    gemv_kernel<<<(3 * H_) / 8, 256, 0, stream>>>(gru_w_ih, xcat,   gru_b_ih, gi, 3 * H_, 2 * H_, 0);
    gemv_kernel<<<(3 * H_) / 8, 256, 0, stream>>>(gru_w_hh, hidden, gru_b_hh, gh, 3 * H_, H_,     0);
    gru_combine_kernel<<<H_ / 256, 256, 0, stream>>>(gi, gh, hidden, o_hnew);

    // out = softmax(lin_w @ h_new + lin_b)
    gemv_kernel<<<OUT_ / 8, 256, 0, stream>>>(lin_w, o_hnew, lin_b, logit, OUT_, H_, 0);
    softmax_kernel<<<1, 1024, 0, stream>>>(logit, o_softmax, OUT_);
}
